// GitLoss_79894981640613
// MI455X (gfx1250) — matrix operand order repaired and checked by reading, not compiled, not run on hardware
//
#include <hip/hip_runtime.h>
#include <hip/hip_bf16.h>

typedef __attribute__((ext_vector_type(16))) __bf16 v16bf;
typedef __attribute__((ext_vector_type(8)))  __bf16 v8bf;
typedef __attribute__((ext_vector_type(8)))  float  v8f;

union BF16x16 { v16bf v; v8bf h[2]; };
static_assert(sizeof(v8bf) == 16, "v8bf must be 16 bytes");

constexpr int BATCH = 4096;
constexpr int NCLS  = 10000;
constexpr int DIM   = 512;

// GEMM tiling: block = 256 threads = 8 waves; each wave owns 16 rows, 8 n-tiles of 16.
constexpr int BM = 128;
constexpr int BN = 128;
constexpr int NTILES = (NCLS + BN - 1) / BN;   // 79 column tiles
constexpr int NCHUNK = 8;                      // column chunks (grid.y), partials per chunk
constexpr int KCHUNKS = DIM / 32;              // 16

// ---- workspace layout (float offsets), bf16 region after ----
constexpr size_t OFF_SUMX   = 0;                                   // NCLS*DIM f32
constexpr size_t OFF_COUNTS = OFF_SUMX + (size_t)NCLS * DIM;       // NCLS f32
constexpr size_t OFF_PART   = OFF_COUNTS + NCLS;                   // NCHUNK*BATCH f32
constexpr size_t OFF_XN     = OFF_PART + (size_t)NCHUNK * BATCH;   // BATCH f32
constexpr size_t OFF_CN     = OFF_XN + BATCH;                      // NCLS f32
constexpr size_t OFF_DIST   = OFF_CN + NCLS;                       // BATCH f32
constexpr size_t OFF_BF16_BYTES = (OFF_DIST + BATCH) * 4;          // 20,723,840 (16B aligned)

__global__ void zero_f32_kernel(float* __restrict__ p, int n) {
  for (int i = blockIdx.x * blockDim.x + threadIdx.x; i < n; i += gridDim.x * blockDim.x)
    p[i] = 0.0f;
}

// One block per sample: scatter into sum_x/counts, emit bf16 hi/lo split of x, and ||x||^2.
__global__ __launch_bounds__(256) void scatter_kernel(
    const float* __restrict__ x, const int* __restrict__ labels,
    float* __restrict__ sum_x, float* __restrict__ counts,
    __bf16* __restrict__ x_hi, __bf16* __restrict__ x_lo,
    float* __restrict__ xn2)
{
  __shared__ float red[256];
  const int i = blockIdx.x;
  const int t = threadIdx.x;
  const int lab = labels[i];
  const size_t base  = (size_t)i   * DIM;
  const size_t cbase = (size_t)lab * DIM;
  float local = 0.0f;
#pragma unroll
  for (int k = t; k < DIM; k += 256) {
    float v = x[base + k];
    atomicAdd(&sum_x[cbase + k], v);
    __bf16 hi = (__bf16)v;
    __bf16 lo = (__bf16)(v - (float)hi);
    x_hi[base + k] = hi;
    x_lo[base + k] = lo;
    local += v * v;
  }
  red[t] = local; __syncthreads();
  for (int s = 128; s > 0; s >>= 1) { if (t < s) red[t] += red[t + s]; __syncthreads(); }
  if (t == 0) { xn2[i] = red[0]; atomicAdd(&counts[lab], 1.0f); }
}

// One block per class: centers_new = a*centers + b*sum_x, bf16 split + exact ||c_new||^2.
__global__ __launch_bounds__(256) void centers_kernel(
    const float* __restrict__ centers, const float* __restrict__ sum_x,
    const float* __restrict__ counts, const float* __restrict__ lr,
    __bf16* __restrict__ c_hi, __bf16* __restrict__ c_lo,
    float* __restrict__ cn2)
{
  __shared__ float red[256];
  const int j = blockIdx.x;
  const int t = threadIdx.x;
  const float cnt = counts[j];
  const float l = lr[0];
  const float a = 1.0f - l * cnt / (1.0f + cnt);
  const float b = l / (1.0f + cnt);
  const size_t base = (size_t)j * DIM;
  float local = 0.0f;
#pragma unroll
  for (int k = t; k < DIM; k += 256) {
    float cv = a * centers[base + k] + b * sum_x[base + k];
    __bf16 hi = (__bf16)cv;
    __bf16 lo = (__bf16)(cv - (float)hi);
    c_hi[base + k] = hi;
    c_lo[base + k] = lo;
    local += cv * cv;
  }
  red[t] = local; __syncthreads();
  for (int s = 128; s > 0; s >>= 1) { if (t < s) red[t] += red[t + s]; __syncthreads(); }
  if (t == 0) cn2[j] = red[0];
}

// Fused WMMA GEMM + 1/(1+d) + per-row partial sums (no atomics: one write per row per chunk).
// B tile double-buffered in LDS (one barrier per K-chunk); WMMAs issued pairwise across two
// n-tiles so consecutive matrix ops never chain on the same accumulator.
__global__ __launch_bounds__(256) void gemm_rowsum_kernel(
    const __bf16* __restrict__ x_hi, const __bf16* __restrict__ x_lo,
    const __bf16* __restrict__ c_hi, const __bf16* __restrict__ c_lo,
    const float* __restrict__ xn2, const float* __restrict__ cn2,
    float* __restrict__ partial)
{
  // 2 buffers x (hi+lo) x 128 cols x 32 K. Row stride padded to 40 bf16 (80 B): spreads
  // LDS banks while keeping 16-B alignment for b128 ds accesses. Total 40,960 B.
  __shared__ __attribute__((aligned(16))) __bf16 sBh[2][BN][40];
  __shared__ __attribute__((aligned(16))) __bf16 sBl[2][BN][40];

  const int tid  = threadIdx.x;
  const int wave = tid >> 5;
  const int lane = tid & 31;
  const int lm   = lane & 15;   // A-row / B-col / D-col within 16
  const int lh   = lane >> 4;   // lane-half selects K sub-block / M-half

  const int mBase = blockIdx.x * BM + wave * 16;
  const int chunk = blockIdx.y;
  const int ct0 = (chunk * NTILES) / NCHUNK;
  const int ct1 = ((chunk + 1) * NTILES) / NCHUNK;

  // ||x||^2 for this lane's 8 output rows: M = v + lh*8 (C/D layout, ISA 7.12.2)
  float xn[8];
#pragma unroll
  for (int v = 0; v < 8; ++v) xn[v] = xn2[mBase + v + lh * 8];

  float acc[8] = {0.f,0.f,0.f,0.f,0.f,0.f,0.f,0.f};

  const __bf16* aRowH = x_hi + (size_t)(mBase + lm) * DIM;
  const __bf16* aRowL = x_lo + (size_t)(mBase + lm) * DIM;
  const int kbA = lh * 16;  // A: lane half covers K = kbA..kbA+15
  const int kbB = lh * 16;  // B: lane half covers K = kbB..kbB+15

  // cooperative B-tile loader mapping (2 threads per column, 16 K each)
  const int col = tid >> 1;
  const int kh  = (tid & 1) * 16;

  for (int ct = ct0; ct < ct1; ++ct) {
    const int nBase = ct * BN;
    int nG = nBase + col; if (nG >= NCLS) nG = NCLS - 1;
    const size_t gRow = (size_t)nG * DIM + kh;   // + k0 per chunk

    v8f c[8] = {};  // 8 n-tiles of 16x16 f32 accumulators

    // ---- prologue: stage K-chunk 0 into buffer 0 ----
    {
      const v8bf h0 = *(const v8bf*)(c_hi + gRow);
      const v8bf h1 = *(const v8bf*)(c_hi + gRow + 8);
      const v8bf l0 = *(const v8bf*)(c_lo + gRow);
      const v8bf l1 = *(const v8bf*)(c_lo + gRow + 8);
      *(v8bf*)&sBh[0][col][kh]     = h0;
      *(v8bf*)&sBh[0][col][kh + 8] = h1;
      *(v8bf*)&sBl[0][col][kh]     = l0;
      *(v8bf*)&sBl[0][col][kh + 8] = l1;
    }
    __syncthreads();

    for (int kc = 0; kc < KCHUNKS; ++kc) {
      const int buf = kc & 1;
      const int k0  = kc * 32;
      const bool more = (kc + 1) < KCHUNKS;

      // ---- issue next chunk's global loads early (latency hidden under WMMAs) ----
      v8bf nh0, nh1, nl0, nl1;
      if (more) {
        const size_t g = gRow + (size_t)(k0 + 32);
        nh0 = *(const v8bf*)(c_hi + g);
        nh1 = *(const v8bf*)(c_hi + g + 8);
        nl0 = *(const v8bf*)(c_lo + g);
        nl1 = *(const v8bf*)(c_lo + g + 8);
      }

      // ---- A fragments (direct global, L2-resident) ----
      BF16x16 aH, aL;
      aH.h[0] = *(const v8bf*)(aRowH + k0 + kbA);
      aH.h[1] = *(const v8bf*)(aRowH + k0 + kbA + 8);
      aL.h[0] = *(const v8bf*)(aRowL + k0 + kbA);
      aL.h[1] = *(const v8bf*)(aRowL + k0 + kbA + 8);

      // ---- compute: two n-tiles at a time, WMMAs interleaved across accumulators ----
#pragma unroll
      for (int nt = 0; nt < 8; nt += 2) {
        BF16x16 b0H, b0L, b1H, b1L;
        const __bf16* p0h = &sBh[buf][nt * 16 + lm][kbB];
        const __bf16* p0l = &sBl[buf][nt * 16 + lm][kbB];
        const __bf16* p1h = &sBh[buf][(nt + 1) * 16 + lm][kbB];
        const __bf16* p1l = &sBl[buf][(nt + 1) * 16 + lm][kbB];
        b0H.h[0] = *(const v8bf*)(p0h); b0H.h[1] = *(const v8bf*)(p0h + 8);
        b0L.h[0] = *(const v8bf*)(p0l); b0L.h[1] = *(const v8bf*)(p0l + 8);
        b1H.h[0] = *(const v8bf*)(p1h); b1H.h[1] = *(const v8bf*)(p1h + 8);
        b1L.h[0] = *(const v8bf*)(p1l); b1L.h[1] = *(const v8bf*)(p1l + 8);
        // f32 dot via bf16 split: hi*hi + hi*lo + lo*hi (lo*lo dropped, ~1e-4 abs on d~1e3)
        c[nt]     = __builtin_amdgcn_wmma_f32_16x16x32_bf16(false, aH.v, false, b0H.v, (short)0, c[nt],     false, false);
        c[nt + 1] = __builtin_amdgcn_wmma_f32_16x16x32_bf16(false, aH.v, false, b1H.v, (short)0, c[nt + 1], false, false);
        c[nt]     = __builtin_amdgcn_wmma_f32_16x16x32_bf16(false, aH.v, false, b0L.v, (short)0, c[nt],     false, false);
        c[nt + 1] = __builtin_amdgcn_wmma_f32_16x16x32_bf16(false, aH.v, false, b1L.v, (short)0, c[nt + 1], false, false);
        c[nt]     = __builtin_amdgcn_wmma_f32_16x16x32_bf16(false, aL.v, false, b0H.v, (short)0, c[nt],     false, false);
        c[nt + 1] = __builtin_amdgcn_wmma_f32_16x16x32_bf16(false, aL.v, false, b1H.v, (short)0, c[nt + 1], false, false);
      }

      // ---- stash next chunk into the other buffer; one barrier per iteration ----
      if (more) {
        const int nb = buf ^ 1;
        *(v8bf*)&sBh[nb][col][kh]     = nh0;
        *(v8bf*)&sBh[nb][col][kh + 8] = nh1;
        *(v8bf*)&sBl[nb][col][kh]     = nl0;
        *(v8bf*)&sBl[nb][col][kh + 8] = nl1;
      }
      __syncthreads();
    }

    // ---- fused epilogue: d = ||x||^2 + ||c||^2 - 2*dot ; s = 1/(1+d) ; mask OOB cols ----
#pragma unroll
    for (int nt = 0; nt < 8; ++nt) {
      const int nGc = nBase + nt * 16 + lm;
      const bool valid = nGc < NCLS;
      const float cn = cn2[valid ? nGc : (NCLS - 1)];
#pragma unroll
      for (int v = 0; v < 8; ++v) {
        float dd = xn[v] + cn - 2.0f * c[nt][v];
        float s  = 1.0f / (1.0f + dd);
        if (valid) acc[v] += s;
      }
    }
  }

  // reduce over the 16 lanes holding one row's columns; lanes 0/16 write partials once
#pragma unroll
  for (int v = 0; v < 8; ++v) {
    float a = acc[v];
    a += __shfl_xor(a, 1);
    a += __shfl_xor(a, 2);
    a += __shfl_xor(a, 4);
    a += __shfl_xor(a, 8);
    if (lm == 0) partial[(size_t)chunk * BATCH + mBase + v + lh * 8] = a;
  }
}

// One wave per sample: recompute own-class entry with the same split math, clamp.
__global__ __launch_bounds__(256) void finalize_kernel(
    const int* __restrict__ labels,
    const __bf16* __restrict__ x_hi, const __bf16* __restrict__ x_lo,
    const __bf16* __restrict__ c_hi, const __bf16* __restrict__ c_lo,
    const float* __restrict__ xn2, const float* __restrict__ cn2,
    const float* __restrict__ partial, float* __restrict__ dist)
{
  const int wave = threadIdx.x >> 5;
  const int lane = threadIdx.x & 31;
  const int i = blockIdx.x * 8 + wave;
  const int lab = labels[i];
  const size_t xb = (size_t)i   * DIM + (size_t)lane * 16;
  const size_t cb = (size_t)lab * DIM + (size_t)lane * 16;
  float dot = 0.0f;
#pragma unroll
  for (int u = 0; u < 16; ++u) {
    float xh = (float)x_hi[xb + u], xl = (float)x_lo[xb + u];
    float ch = (float)c_hi[cb + u], cl = (float)c_lo[cb + u];
    dot += xh * ch + xh * cl + xl * ch;
  }
  dot += __shfl_xor(dot, 1);
  dot += __shfl_xor(dot, 2);
  dot += __shfl_xor(dot, 4);
  dot += __shfl_xor(dot, 8);
  dot += __shfl_xor(dot, 16);
  if (lane == 0) {
    float dd  = xn2[i] + cn2[lab] - 2.0f * dot;
    float own = 1.0f / (1.0f + dd);
    float rs = 0.0f;
#pragma unroll
    for (int cch = 0; cch < NCHUNK; ++cch) rs += partial[(size_t)cch * BATCH + i];
    float v = rs - own;
    v = fminf(fmaxf(v, 1e-12f), 1e12f);
    dist[i] = v;
  }
}

// Deterministic single-block mean.
__global__ __launch_bounds__(256) void reduce_mean_kernel(const float* __restrict__ dist,
                                                          float* __restrict__ out) {
  __shared__ float red[256];
  float s = 0.0f;
  for (int i = threadIdx.x; i < BATCH; i += 256) s += dist[i];
  red[threadIdx.x] = s; __syncthreads();
  for (int st = 128; st > 0; st >>= 1) {
    if (threadIdx.x < st) red[threadIdx.x] += red[threadIdx.x + st];
    __syncthreads();
  }
  if (threadIdx.x == 0) out[0] = red[0] * (1.0f / (float)BATCH);
}

extern "C" void kernel_launch(void* const* d_in, const int* in_sizes, int n_in,
                              void* d_out, int out_size, void* d_ws, size_t ws_size,
                              hipStream_t stream) {
  (void)in_sizes; (void)n_in; (void)out_size; (void)ws_size;
  const float* x       = (const float*)d_in[0];
  const int*   labels  = (const int*)d_in[1];
  const float* centers = (const float*)d_in[2];
  const float* lr      = (const float*)d_in[3];
  float* out = (float*)d_out;

  float* wsf     = (float*)d_ws;
  float* sum_x   = wsf + OFF_SUMX;
  float* counts  = wsf + OFF_COUNTS;
  float* partial = wsf + OFF_PART;
  float* xn2     = wsf + OFF_XN;
  float* cn2     = wsf + OFF_CN;
  float* dist    = wsf + OFF_DIST;
  __bf16* x_hi = (__bf16*)((char*)d_ws + OFF_BF16_BYTES);
  __bf16* x_lo = x_hi + (size_t)BATCH * DIM;
  __bf16* c_hi = x_lo + (size_t)BATCH * DIM;
  __bf16* c_lo = c_hi + (size_t)NCLS * DIM;

  const int nZero = NCLS * DIM + NCLS;   // sum_x + counts (contiguous); re-zeroed every call
  zero_f32_kernel<<<2048, 256, 0, stream>>>(sum_x, nZero);
  scatter_kernel<<<BATCH, 256, 0, stream>>>(x, labels, sum_x, counts, x_hi, x_lo, xn2);
  centers_kernel<<<NCLS, 256, 0, stream>>>(centers, sum_x, counts, lr, c_hi, c_lo, cn2);
  gemm_rowsum_kernel<<<dim3(BATCH / BM, NCHUNK), 256, 0, stream>>>(x_hi, x_lo, c_hi, c_lo,
                                                                   xn2, cn2, partial);
  finalize_kernel<<<BATCH / 8, 256, 0, stream>>>(labels, x_hi, x_lo, c_hi, c_lo,
                                                 xn2, cn2, partial, dist);
  reduce_mean_kernel<<<1, 256, 0, stream>>>(dist, out);
}
